// ParticleNet_21844203668002
// MI455X (gfx1250) — matrix operand order repaired and checked by reading, not compiled, not run on hardware
//
#include <hip/hip_runtime.h>
#include <hip/hip_bf16.h>

// ---------------------------------------------------------------------------
// Problem constants (from reference)
// ---------------------------------------------------------------------------
#define NNODES   30000
#define NEDGES   480000
#define NFEAT    16
#define NHID     128
#define NCLS     2
#define NGRAPHS  256
#define GN_EPS   1e-5f

typedef __attribute__((ext_vector_type(16))) __bf16 v16bf;
typedef __attribute__((ext_vector_type(8)))  float  v8f;

// ---------------------------------------------------------------------------
// Small device helpers
// ---------------------------------------------------------------------------

// Branchless SELU: scale*max(x,0) + scale*alpha*(exp(min(x,0)) - 1)
//  x>0: second term = scale*alpha*(e^0-1) = 0;  x<=0: first term = 0.
__device__ __forceinline__ float selu_f(float x) {
    const float sa = 1.7580993408473766f;   // scale*alpha
    const float sc = 1.0507009873554805f;   // scale
    return fmaxf(x, 0.f) * sc + sa * (__expf(fminf(x, 0.f)) - 1.f);
}

// Monotonic total-order mapping float -> u32 (branchless):
//   neg: ~bits ; pos: bits | 0x80000000.  atomicMax(u32) == float max.
__device__ __forceinline__ unsigned int float_enc(float f) {
    unsigned int u = __float_as_uint(f);
    unsigned int mask = (unsigned int)(((int)u) >> 31);   // all-ones if negative
    return u ^ (mask | 0x80000000u);
}
__device__ __forceinline__ float float_dec(unsigned int e) {
    unsigned int mask = (unsigned int)(~(((int)e) >> 31)); // 0 if top bit set
    return __uint_as_float(e ^ (mask | 0x80000000u));
}
#define ENC_NEG_INF 0x007FFFFFu   // float_enc(-inf)

// ---------------------------------------------------------------------------
// Wave-level GEMM tile: D[16,64-col-slice] = A[16,K] * W[K,128] (+bias, epi)
//   A: LDS, row-major [64][K] bf16  (this wave reads rows m0..m0+15)
//   W: global, row-major [K][128] bf16
// Each wave owns: m-subtile = wave>>1, n-half = wave&1 (4 n-subtiles of 16).
// MODE 0: selu -> OutLds (bf16 [64][128]);  MODE 1: atomic segment-max -> agg
// ---------------------------------------------------------------------------
template<int K, int MODE>
__device__ __forceinline__ void wave_gemm(
    const __bf16* __restrict__ A,
    const __bf16* __restrict__ W,
    const float*  __restrict__ bias,
    __bf16*       __restrict__ OutLds,
    unsigned int* __restrict__ agg,     // encoded-f32 when MODE==1
    const int*    __restrict__ dstS,
    int tid)
{
    const int lane = tid & 31;
    const int wave = tid >> 5;
    const int m0   = (wave >> 1) * 16;   // 4 m-subtiles over 64 edges
    const int nb   = (wave & 1) * 4;     // 4 n-subtiles per wave
    const int mrow = lane & 15;
    const int hi   = lane >> 4;

    v8f acc[4];
#pragma unroll
    for (int n = 0; n < 4; ++n)
        acc[n] = v8f{0.f,0.f,0.f,0.f,0.f,0.f,0.f,0.f};

#pragma unroll
    for (int k0 = 0; k0 < K; k0 += 32) {
        // A fragment (ISA 16-bit A 16x32 layout): lane holds row m0+mrow,
        // halves: K = k0+8*hi+e (e<8) and K = k0+8*hi+8+e (e>=8)
        const __bf16* arow = A + (m0 + mrow) * K + k0 + 8 * hi;
        v16bf a;
#pragma unroll
        for (int e = 0; e < 8; ++e) { a[e] = arow[e]; a[e + 8] = arow[e + 16]; }

#pragma unroll
        for (int n = 0; n < 4; ++n) {
            // B fragment: lane holds N-col (lane&15), K = k0+8*hi+e (e<8) and K = k0+8*hi+8+e (e>=8)
            const __bf16* bcol = W + (k0 + 8 * hi) * NHID + (nb + n) * 16 + mrow;
            v16bf b;
#pragma unroll
            for (int e = 0; e < 8; ++e) { b[e] = bcol[e * NHID]; b[e + 8] = bcol[(e + 16) * NHID]; }
            acc[n] = __builtin_amdgcn_wmma_f32_16x16x32_bf16(
                false, a, false, b, (short)0, acc[n], false, false);
        }
    }

    // Epilogue: C/D layout -> lane L, vgpr r: M = r + 8*hi, N = L&15
#pragma unroll
    for (int n = 0; n < 4; ++n) {
        const int col = (nb + n) * 16 + (lane & 15);
        const float bb = bias[col];
#pragma unroll
        for (int r = 0; r < 8; ++r) {
            float v = acc[n][r] + bb;
            const int m = m0 + 8 * hi + r;
            if (MODE == 0) {
                OutLds[m * NHID + col] = (__bf16)selu_f(v);
            } else {
                atomicMax(&agg[dstS[m] * NHID + col], float_enc(v));
            }
        }
    }
}

// ---------------------------------------------------------------------------
// Fused EdgeConv MLP over a tile of 64 edges:
//   t = [x_i, x_j - x_i] ; h = selu(t@W1+b1) ; h = selu(h@W2+b2) ; h = h@W3+b3
//   agg[dst] = max(agg[dst], h)   (agg pre-initialized to enc(-inf))
// ---------------------------------------------------------------------------
template<int CIN>
__global__ __launch_bounds__(256) void edgeconv_kernel(
    const float*     __restrict__ x,      // [NNODES, CIN] f32
    const long long* __restrict__ src,    // [NEDGES]
    const long long* __restrict__ dst,    // [NEDGES]
    const __bf16* __restrict__ W1, const float* __restrict__ b1,  // [2C,128]
    const __bf16* __restrict__ W2, const float* __restrict__ b2,  // [128,128]
    const __bf16* __restrict__ W3, const float* __restrict__ b3,  // [128,128]
    unsigned int* __restrict__ agg)       // [NNODES, 128] encoded, init enc(-inf)
{
    constexpr int K1 = 2 * CIN;
    constexpr int R0 = 64 * (K1 > NHID ? K1 : NHID);   // region for t (and later h2)
    __shared__ __align__(16) __bf16 smem[R0 + 64 * NHID];
    __shared__ int dstS[64];

    __bf16* tT = smem;            // [64][K1]
    __bf16* hA = smem + R0;       // [64][128]
    __bf16* hB = smem;            // aliases tT (dead after GEMM1)

    const int tid = threadIdx.x;
    const int e0  = blockIdx.x * 64;

    // ---- gather stage: 4 threads per edge ----
    {
        const int e    = tid >> 2;
        const int part = tid & 3;
        const int d = (int)dst[e0 + e];
        const int s = (int)src[e0 + e];
        if (part == 0) dstS[e] = d;
        constexpr int CH = CIN / 4;
#pragma unroll
        for (int cc = 0; cc < CH; ++cc) {
            const int c = part * CH + cc;
            const float xi = x[d * CIN + c];
            const float xj = x[s * CIN + c];
            tT[e * K1 + c]       = (__bf16)xi;
            tT[e * K1 + CIN + c] = (__bf16)(xj - xi);
        }
    }
    __syncthreads();

    wave_gemm<K1,   0>(tT, W1, b1, hA,      nullptr, nullptr, tid);
    __syncthreads();
    wave_gemm<NHID, 0>(hA, W2, b2, hB,      nullptr, nullptr, tid);
    __syncthreads();
    wave_gemm<NHID, 1>(hB, W3, b3, nullptr, agg,     dstS,    tid);
}

// ---------------------------------------------------------------------------
// Elementwise / reduction helper kernels
// ---------------------------------------------------------------------------
__global__ void fill_kernel(float* __restrict__ p, float v, int n) {
    int i = blockIdx.x * blockDim.x + threadIdx.x;
    if (i < n) p[i] = v;
}

__global__ void fill_u32_kernel(unsigned int* __restrict__ p, unsigned int v, int n) {
    int i = blockIdx.x * blockDim.x + threadIdx.x;
    if (i < n) p[i] = v;
}

__global__ void f32_to_bf16_kernel(const float* __restrict__ in,
                                   __bf16* __restrict__ out, int n) {
    int i = blockIdx.x * blockDim.x + threadIdx.x;
    if (i < n) out[i] = (__bf16)in[i];
}

__global__ void gn_count_kernel(const long long* __restrict__ batch,
                                float* __restrict__ cnt, int n) {
    int i = blockIdx.x * blockDim.x + threadIdx.x;
    if (i < n) atomicAdd(&cnt[(int)batch[i]], 1.0f);
}

template<int F>
__global__ void gn_sum_kernel(const float* __restrict__ x,
                              const long long* __restrict__ batch,
                              float* __restrict__ sum) {
    int i = blockIdx.x * blockDim.x + threadIdx.x;
    if (i < NNODES * F) {
        int node = i / F, f = i % F;
        atomicAdd(&sum[(int)batch[node] * F + f], x[i]);
    }
}

template<int F>
__global__ void gn_center_kernel(const float* __restrict__ x,
                                 const long long* __restrict__ batch,
                                 const float* __restrict__ sum,
                                 const float* __restrict__ cnt,
                                 const float* __restrict__ mscale,
                                 float* __restrict__ out,
                                 float* __restrict__ varsum) {
    int i = blockIdx.x * blockDim.x + threadIdx.x;
    if (i < NNODES * F) {
        int node = i / F, f = i % F;
        int g = (int)batch[node];
        float mean = sum[g * F + f] / fmaxf(cnt[g], 1.f);
        float o = x[i] - mean * mscale[f];
        out[i] = o;
        atomicAdd(&varsum[g * F + f], o * o);
    }
}

template<int F>
__global__ void gn_final_kernel(const long long* __restrict__ batch,
                                const float* __restrict__ varsum,
                                const float* __restrict__ cnt,
                                const float* __restrict__ weight,
                                const float* __restrict__ bias,
                                float* __restrict__ out) {
    int i = blockIdx.x * blockDim.x + threadIdx.x;
    if (i < NNODES * F) {
        int node = i / F, f = i % F;
        int g = (int)batch[node];
        float var = varsum[g * F + f] / fmaxf(cnt[g], 1.f);
        out[i] = weight[f] * out[i] * rsqrtf(var + GN_EPS) + bias[f];
    }
}

// decode(agg) -> relu(where(isfinite, v, 0))
__global__ void agg_finish_kernel(const unsigned int* __restrict__ agg,
                                  float* __restrict__ out, int n) {
    int i = blockIdx.x * blockDim.x + threadIdx.x;
    if (i < n) {
        float v = float_dec(agg[i]);
        if (v != v) v = 0.f;                               // NaN
        if (v == __builtin_inff() || v == -__builtin_inff()) v = 0.f;
        out[i] = fmaxf(v, 0.f);
    }
}

__global__ void pool_sum_kernel(const float* __restrict__ h,
                                const long long* __restrict__ batch,
                                float* __restrict__ pool, int n) {
    int i = blockIdx.x * blockDim.x + threadIdx.x;
    if (i < n) {
        int node = i >> 7;          // /128
        int f    = i & 127;
        atomicAdd(&pool[(int)batch[node] * NHID + f], h[i]);
    }
}

// One block (128 threads) per graph: mean-pool -> dense+relu -> out -> softmax
__global__ __launch_bounds__(128) void head_kernel(
    const float* __restrict__ pool, const float* __restrict__ cnt,
    const float* __restrict__ dW, const float* __restrict__ db,
    const float* __restrict__ oW, const float* __restrict__ ob,
    float* __restrict__ out)
{
    __shared__ float gbuf[NHID];
    __shared__ float dbuf[NHID];
    __shared__ float logits[NCLS];
    const int g = blockIdx.x, f = threadIdx.x;

    gbuf[f] = pool[g * NHID + f] / fmaxf(cnt[g], 1.f);
    __syncthreads();

    float acc = db[f];
    for (int k = 0; k < NHID; ++k) acc += gbuf[k] * dW[k * NHID + f];
    dbuf[f] = fmaxf(acc, 0.f);
    __syncthreads();

    if (f < NCLS) {
        float a = ob[f];
        for (int k = 0; k < NHID; ++k) a += dbuf[k] * oW[k * NCLS + f];
        logits[f] = a;
    }
    __syncthreads();

    if (f == 0) {
        float m  = fmaxf(logits[0], logits[1]);
        float e0 = __expf(logits[0] - m), e1 = __expf(logits[1] - m);
        float inv = 1.f / (e0 + e1);
        out[g * NCLS + 0] = e0 * inv;
        out[g * NCLS + 1] = e1 * inv;
    }
}

// ---------------------------------------------------------------------------
// Host orchestration
// ---------------------------------------------------------------------------
template<int F>
static void run_graph_norm(const float* xin, float* xout,
                           const long long* batch,
                           const float* weight, const float* bias, const float* mscale,
                           float* sum, float* var, const float* cnt,
                           hipStream_t s)
{
    const int n  = NNODES * F;
    const int gS = (NGRAPHS * F + 255) / 256;
    const int gN = (n + 255) / 256;
    fill_kernel<<<gS, 256, 0, s>>>(sum, 0.f, NGRAPHS * F);
    fill_kernel<<<gS, 256, 0, s>>>(var, 0.f, NGRAPHS * F);
    gn_sum_kernel<F><<<gN, 256, 0, s>>>(xin, batch, sum);
    gn_center_kernel<F><<<gN, 256, 0, s>>>(xin, batch, sum, cnt, mscale, xout, var);
    gn_final_kernel<F><<<gN, 256, 0, s>>>(batch, var, cnt, weight, bias, xout);
}

extern "C" void kernel_launch(void* const* d_in, const int* in_sizes, int n_in,
                              void* d_out, int out_size, void* d_ws, size_t ws_size,
                              hipStream_t stream)
{
    (void)in_sizes; (void)n_in; (void)out_size; (void)ws_size;

    // ---- inputs, setup_inputs() dict insertion order (nested dicts too) ----
    const float*     x     = (const float*)d_in[0];                // [30000,16]
    const long long* eidx  = (const long long*)d_in[1];            // [2,480000]
    const long long* batch = (const long long*)d_in[2];            // [30000]
    const long long* src   = eidx;
    const long long* dst   = eidx + NEDGES;

    // params: gn0(w,b,ms) gn1 gn2 conv1(W1,b1,W2,b2,W3,b3) conv2 conv3 dW db oW ob
    const float* gnW[3], *gnB[3], *gnM[3];
    for (int l = 0; l < 3; ++l) {
        gnW[l] = (const float*)d_in[3 + 3 * l + 0];
        gnB[l] = (const float*)d_in[3 + 3 * l + 1];
        gnM[l] = (const float*)d_in[3 + 3 * l + 2];
    }
    const float* cW1[3], *cb1[3], *cW2[3], *cb2[3], *cW3[3], *cb3[3];
    for (int l = 0; l < 3; ++l) {
        const int base = 12 + 6 * l;
        cW1[l] = (const float*)d_in[base + 0];
        cb1[l] = (const float*)d_in[base + 1];
        cW2[l] = (const float*)d_in[base + 2];
        cb2[l] = (const float*)d_in[base + 3];
        cW3[l] = (const float*)d_in[base + 4];
        cb3[l] = (const float*)d_in[base + 5];
    }
    const float* denseW = (const float*)d_in[30];
    const float* denseB = (const float*)d_in[31];
    const float* outW   = (const float*)d_in[32];
    const float* outB   = (const float*)d_in[33];

    // ---- workspace carve-out ----
    char*  w = (char*)d_ws;
    size_t o = 0;
    auto carve = [&](size_t bytes) -> void* {
        void* p = w + o;
        o += (bytes + 511) & ~(size_t)511;
        return p;
    };
    float*        X    = (float*)carve((size_t)NNODES * NHID * 4);  // activations ping
    float*        Y    = (float*)carve((size_t)NNODES * NHID * 4);  // activations pong
    unsigned int* AGG  = (unsigned int*)carve((size_t)NNODES * NHID * 4);
    float*        SUM  = (float*)carve((size_t)NGRAPHS * NHID * 4);
    float*        VAR  = (float*)carve((size_t)NGRAPHS * NHID * 4);
    float*        CNT  = (float*)carve((size_t)NGRAPHS * 4);
    float*        POOL = (float*)carve((size_t)NGRAPHS * NHID * 4);
    __bf16*       W1bf = (__bf16*)carve((size_t)2 * NHID * NHID * 2); // up to [256,128]
    __bf16*       W2bf = (__bf16*)carve((size_t)NHID * NHID * 2);
    __bf16*       W3bf = (__bf16*)carve((size_t)NHID * NHID * 2);

    const int nHid   = NNODES * NHID;
    const int gHid   = (nHid + 255) / 256;
    const int gEdges = NEDGES / 64;                 // 7500, exact

    // per-graph node counts (shared by all norms + mean pool)
    fill_kernel<<<1, 256, 0, stream>>>(CNT, 0.f, NGRAPHS);
    gn_count_kernel<<<(NNODES + 255) / 256, 256, 0, stream>>>(batch, CNT, NNODES);

    // ---- layer 0: graph_norm(x) over 16 features -> Y[:,0:16] ----
    run_graph_norm<NFEAT>(x, Y, batch, gnW[0], gnB[0], gnM[0], SUM, VAR, CNT, stream);

    // ---- conv1 (CIN=16) ----
    f32_to_bf16_kernel<<<(2*NFEAT*NHID + 255)/256, 256, 0, stream>>>(cW1[0], W1bf, 2*NFEAT*NHID);
    f32_to_bf16_kernel<<<(NHID*NHID + 255)/256,   256, 0, stream>>>(cW2[0], W2bf, NHID*NHID);
    f32_to_bf16_kernel<<<(NHID*NHID + 255)/256,   256, 0, stream>>>(cW3[0], W3bf, NHID*NHID);
    fill_u32_kernel<<<gHid, 256, 0, stream>>>(AGG, ENC_NEG_INF, nHid);
    edgeconv_kernel<NFEAT><<<gEdges, 256, 0, stream>>>(Y, src, dst,
        W1bf, cb1[0], W2bf, cb2[0], W3bf, cb3[0], AGG);
    agg_finish_kernel<<<gHid, 256, 0, stream>>>(AGG, X, nHid);

    // ---- gn1 -> conv2 ----
    run_graph_norm<NHID>(X, Y, batch, gnW[1], gnB[1], gnM[1], SUM, VAR, CNT, stream);
    f32_to_bf16_kernel<<<(2*NHID*NHID + 255)/256, 256, 0, stream>>>(cW1[1], W1bf, 2*NHID*NHID);
    f32_to_bf16_kernel<<<(NHID*NHID + 255)/256,   256, 0, stream>>>(cW2[1], W2bf, NHID*NHID);
    f32_to_bf16_kernel<<<(NHID*NHID + 255)/256,   256, 0, stream>>>(cW3[1], W3bf, NHID*NHID);
    fill_u32_kernel<<<gHid, 256, 0, stream>>>(AGG, ENC_NEG_INF, nHid);
    edgeconv_kernel<NHID><<<gEdges, 256, 0, stream>>>(Y, src, dst,
        W1bf, cb1[1], W2bf, cb2[1], W3bf, cb3[1], AGG);
    agg_finish_kernel<<<gHid, 256, 0, stream>>>(AGG, X, nHid);

    // ---- gn2 -> conv3 ----
    run_graph_norm<NHID>(X, Y, batch, gnW[2], gnB[2], gnM[2], SUM, VAR, CNT, stream);
    f32_to_bf16_kernel<<<(2*NHID*NHID + 255)/256, 256, 0, stream>>>(cW1[2], W1bf, 2*NHID*NHID);
    f32_to_bf16_kernel<<<(NHID*NHID + 255)/256,   256, 0, stream>>>(cW2[2], W2bf, NHID*NHID);
    f32_to_bf16_kernel<<<(NHID*NHID + 255)/256,   256, 0, stream>>>(cW3[2], W3bf, NHID*NHID);
    fill_u32_kernel<<<gHid, 256, 0, stream>>>(AGG, ENC_NEG_INF, nHid);
    edgeconv_kernel<NHID><<<gEdges, 256, 0, stream>>>(Y, src, dst,
        W1bf, cb1[2], W2bf, cb2[2], W3bf, cb3[2], AGG);
    agg_finish_kernel<<<gHid, 256, 0, stream>>>(AGG, X, nHid);

    // ---- mean pool + head + softmax ----
    fill_kernel<<<(NGRAPHS*NHID + 255)/256, 256, 0, stream>>>(POOL, 0.f, NGRAPHS*NHID);
    pool_sum_kernel<<<gHid, 256, 0, stream>>>(X, batch, POOL, nHid);
    head_kernel<<<NGRAPHS, NHID, 0, stream>>>(POOL, CNT, denseW, denseB, outW, outB,
                                              (float*)d_out);
}
